// MultiHeadAttention_66365834658290
// MI455X (gfx1250) — matrix operand order repaired and checked by reading, not compiled, not run on hardware
//
#include <hip/hip_runtime.h>
#include <hip/hip_bf16.h>

typedef __bf16 bf16_t;
typedef __attribute__((ext_vector_type(16))) __bf16 v16bf;
typedef __attribute__((ext_vector_type(8)))  __bf16 v8bf;
typedef __attribute__((ext_vector_type(8)))  float  v8f;

#define WMMA_BF16(a, b, c) \
  __builtin_amdgcn_wmma_f32_16x16x32_bf16(false, (a), false, (b), (short)0, (c), false, false)

static constexpr int Bb = 2;
static constexpr int S  = 2048;
static constexpr int D  = 1024;
static constexpr int H  = 16;
static constexpr int DK = 64;
static constexpr int MT = Bb * S;   // 4096 total rows

__device__ __forceinline__ bf16_t f2bf(float f) {
  unsigned u = __builtin_bit_cast(unsigned, f);
  unsigned r = (u + 0x7FFFu + ((u >> 16) & 1u)) >> 16;
  unsigned short s = (unsigned short)r;
  return __builtin_bit_cast(bf16_t, s);
}

__device__ __forceinline__ v8f vzero8() {
  v8f z;
#pragma unroll
  for (int i = 0; i < 8; ++i) z[i] = 0.0f;
  return z;
}

// LDS byte offset of a generic pointer to __shared__ (ISA 10.2: LDS aperture
// uses addr[31:0] as the offset from the wave's LDS base).
__device__ __forceinline__ unsigned lds_off(const void* p) {
  return (unsigned)(unsigned long long)p;
}

// async global->LDS copy of 32 bytes (two b128), CDNA5 ASYNCcnt path
__device__ __forceinline__ void async_copy32(unsigned dst_lds, const void* gsrc) {
  asm volatile("global_load_async_to_lds_b128 %0, %1, off\n\t"
               "global_load_async_to_lds_b128 %0, %1, off offset:16"
               :: "v"(dst_lds), "v"(gsrc) : "memory");
}
__device__ __forceinline__ void wait_async0() {
  asm volatile("s_wait_asynccnt 0x0" ::: "memory");
}

// ---- elementwise fp32 -> bf16 convert (bandwidth-bound, one-time) ----
__global__ __launch_bounds__(256) void cvt_bf16_kernel(const float* __restrict__ X,
                                                       bf16_t* __restrict__ Y) {
  size_t i = ((size_t)blockIdx.x * 256 + threadIdx.x) * 8;
  float4 a = *(const float4*)(X + i);
  float4 b = *(const float4*)(X + i + 4);
  v8bf o;
  o[0] = f2bf(a.x); o[1] = f2bf(a.y); o[2] = f2bf(a.z); o[3] = f2bf(a.w);
  o[4] = f2bf(b.x); o[5] = f2bf(b.y); o[6] = f2bf(b.z); o[7] = f2bf(b.w);
  *(v8bf*)(Y + i) = o;
}

// ---- fragment loaders (contiguous k order: lane half h holds k = 16h..16h+15) ----

// A 16x32 bf16 from bf16 row-major
__device__ __forceinline__ v16bf load_a_bf(const bf16_t* __restrict__ X, int ld,
                                           int row0, int k0, int lane) {
  int r = row0 + (lane & 15);
  int koff = (lane >> 4) * 16;
  const bf16_t* p = X + (size_t)r * ld + k0 + koff;
  v16bf a;
#pragma unroll
  for (int t = 0; t < 8; ++t) a[t] = p[t];
#pragma unroll
  for (int t = 0; t < 8; ++t) a[8 + t] = p[8 + t];
  return a;
}

// B 32x16 from bf16 source (K contiguous per column; column n = row n of Bp)
__device__ __forceinline__ v16bf load_b_bf(const bf16_t* __restrict__ Bp, int ld,
                                           int n0, int k0, int lane) {
  int n = n0 + (lane & 15);
  int kb = k0 + (lane >> 4) * 16;
  const bf16_t* p = Bp + (size_t)n * ld + kb;
  v16bf b;
#pragma unroll
  for (int t = 0; t < 16; ++t) b[t] = p[t];
  return b;
}

// B 32x16 fragment from an LDS tile with row pitch `pitch` (elements);
// column n = LDS row (n0 + lane%16), K contiguous within the row at k0.
__device__ __forceinline__ v16bf load_b_lds(const bf16_t* base, int pitch,
                                            int n0, int k0, int lane) {
  const bf16_t* p = base + (size_t)(n0 + (lane & 15)) * pitch + k0 + (lane >> 4) * 16;
  v16bf b;
#pragma unroll
  for (int t = 0; t < 16; ++t) b[t] = p[t];
  return b;
}

__device__ __forceinline__ float rmax16(float v) {
  v = fmaxf(v, __shfl_xor(v, 1, 32));
  v = fmaxf(v, __shfl_xor(v, 2, 32));
  v = fmaxf(v, __shfl_xor(v, 4, 32));
  v = fmaxf(v, __shfl_xor(v, 8, 32));
  return v;
}
__device__ __forceinline__ float rsum16(float v) {
  v += __shfl_xor(v, 1, 32);
  v += __shfl_xor(v, 2, 32);
  v += __shfl_xor(v, 4, 32);
  v += __shfl_xor(v, 8, 32);
  return v;
}

// ---- kernel 1: Y = X @ W^T + bias (all bf16 in), store bf16.
// Per wave: 32M x 64N tile. mode 0: [B,H,S,DK] (Q,K). mode 1: [B,H,DK,S] (V^T).
__global__ __launch_bounds__(256) void proj_qkv_kernel(
    const bf16_t* __restrict__ X, const bf16_t* __restrict__ W,
    const float* __restrict__ bias, bf16_t* __restrict__ Y, int mode) {
  const int lane = threadIdx.x & 31;
  const int wgid = blockIdx.x * (blockDim.x >> 5) + (threadIdx.x >> 5);
  const int m0 = (wgid >> 4) * 32;     // 128 m-tiles of 32 rows
  const int n0 = (wgid & 15) * 64;     // 16 n-groups of 64 cols
  const int g = lane >> 4;

  v8f acc0[4] = {vzero8(), vzero8(), vzero8(), vzero8()};
  v8f acc1[4] = {vzero8(), vzero8(), vzero8(), vzero8()};
  for (int k0 = 0; k0 < D; k0 += 32) {
    if (k0 + 32 < D) {
      __builtin_prefetch(X + (size_t)(m0 + (lane & 31)) * D + k0 + 32, 0, 3);
      __builtin_prefetch(W + (size_t)(n0 + (lane & 31)) * D + k0 + 32, 0, 3);
    }
    v16bf a0 = load_a_bf(X, D, m0, k0, lane);
    v16bf a1 = load_a_bf(X, D, m0 + 16, k0, lane);
#pragma unroll
    for (int nt = 0; nt < 4; ++nt) {
      v16bf bw = load_b_bf(W, D, n0 + nt * 16, k0, lane);
      acc0[nt] = WMMA_BF16(a0, bw, acc0[nt]);
      acc1[nt] = WMMA_BF16(a1, bw, acc1[nt]);
    }
  }
#pragma unroll
  for (int nt = 0; nt < 4; ++nt) {
    int n = n0 + nt * 16 + (lane & 15);
    float bval = bias[n];
    int hh = n >> 6, dk = n & 63;
#pragma unroll
    for (int half = 0; half < 2; ++half) {
      const v8f& acc = half ? acc1[nt] : acc0[nt];
#pragma unroll
      for (int j = 0; j < 8; ++j) {
        int m = m0 + half * 16 + j + 8 * g;
        int bb = m >> 11, ss = m & (S - 1);
        float val = acc[j] + bval;
        size_t idx = (mode == 0)
            ? ((((size_t)(bb * H + hh)) * S + ss) * DK + dk)
            : ((((size_t)(bb * H + hh)) * DK + dk) * S + ss);
        Y[idx] = f2bf(val);
      }
    }
  }
}

// ---- kernel 2: flash attention. Block = 4 waves = 4 q-tiles of ONE (b,h).
// K/V chunks (32 keys) staged into LDS via async global->LDS, shared by all waves.
__global__ __launch_bounds__(128) void attn_kernel(
    const bf16_t* __restrict__ Qh, const bf16_t* __restrict__ Kh,
    const bf16_t* __restrict__ Vt, bf16_t* __restrict__ Ao) {
  constexpr int KP = 72;  // K tile pitch: 32 rows(keys) x 64 dk, 144B rows
  constexpr int VP = 40;  // V tile pitch: 64 rows(dk) x 32 keys, 80B rows
  constexpr int PP = 40;  // per-wave P staging pitch
  __shared__ bf16_t smK[32 * KP];
  __shared__ bf16_t smV[64 * VP];
  __shared__ bf16_t smP[4][16 * PP];

  const int tid = threadIdx.x;
  const int lane = tid & 31;
  const int widx = tid >> 5;
  const int b = blockIdx.x >> 9;              // 512 blocks per batch
  const int h = (blockIdx.x >> 5) & 15;       // 32 blocks per head
  const int q0 = (blockIdx.x & 31) * 64 + widx * 16;
  const int g = lane >> 4;
  const size_t head = (size_t)(b * H + h);
  const bf16_t* Qhh = Qh + head * (size_t)S * DK;
  const bf16_t* Khh = Kh + head * (size_t)S * DK;
  const bf16_t* Vth = Vt + head * (size_t)DK * S;
  bf16_t* sm = smP[widx];

  // cooperative-load assignments (128 threads): 32B segments
  const int krow = tid >> 2, kchunk = tid & 3;       // K: 32 rows x 4 chunks
  const int vrow = tid >> 1, vchunk = tid & 1;       // V: 64 rows x 2 chunks
  const unsigned kdst = lds_off(smK + krow * KP + kchunk * 16);
  const unsigned vdst = lds_off(smV + vrow * VP + vchunk * 16);

  const v16bf aq0 = load_a_bf(Qhh, DK, q0, 0, lane);
  const v16bf aq1 = load_a_bf(Qhh, DK, q0, 32, lane);

  v8f acc[4] = {vzero8(), vzero8(), vzero8(), vzero8()};
  float m_run[8], l_run[8];
#pragma unroll
  for (int j = 0; j < 8; ++j) { m_run[j] = -1e30f; l_run[j] = 0.0f; }

  for (int kc = 0; kc < S; kc += 32) {
    __syncthreads();  // previous chunk fully consumed
    async_copy32(kdst, Khh + (size_t)(kc + krow) * DK + kchunk * 16);
    async_copy32(vdst, Vth + (size_t)vrow * S + kc + vchunk * 16);
    wait_async0();
    __syncthreads();  // chunk visible to all waves

    // scores for 32 keys: two 16-key sub-tiles, Dk=64 over two K=32 steps
    v8f s0 = vzero8(), s1 = vzero8();
    s0 = WMMA_BF16(aq0, load_b_lds(smK, KP, 0, 0, lane), s0);
    s0 = WMMA_BF16(aq1, load_b_lds(smK, KP, 0, 32, lane), s0);
    s1 = WMMA_BF16(aq0, load_b_lds(smK, KP, 16, 0, lane), s1);
    s1 = WMMA_BF16(aq1, load_b_lds(smK, KP, 16, 32, lane), s1);

#pragma unroll
    for (int j = 0; j < 8; ++j) { s0[j] *= 0.125f; s1[j] *= 0.125f; }

#pragma unroll
    for (int j = 0; j < 8; ++j) {
      float mt = rmax16(fmaxf(s0[j], s1[j]));
      float mn = fmaxf(m_run[j], mt);
      float corr = __expf(m_run[j] - mn);
      float p0 = __expf(s0[j] - mn);
      float p1 = __expf(s1[j] - mn);
      l_run[j] = l_run[j] * corr + rsum16(p0 + p1);
      m_run[j] = mn;
#pragma unroll
      for (int nt = 0; nt < 4; ++nt) acc[nt][j] *= corr;
      int row = j + 8 * g;
      sm[row * PP + (lane & 15)] = f2bf(p0);
      sm[row * PP + 16 + (lane & 15)] = f2bf(p1);
    }

    // reload P as A-layout 16x32 fragment (per-wave buffer, same-wave ordering)
    v16bf pa;
    {
      const bf16_t* lp = sm + (lane & 15) * PP + g * 16;
#pragma unroll
      for (int t = 0; t < 8; ++t) pa[t] = lp[t];
#pragma unroll
      for (int t = 0; t < 8; ++t) pa[8 + t] = lp[8 + t];
    }

    // out += P @ V : B columns = dk rows of smV, K = 32 keys contiguous
#pragma unroll
    for (int nt = 0; nt < 4; ++nt) {
      v16bf bv = load_b_lds(smV, VP, nt * 16, 0, lane);
      acc[nt] = WMMA_BF16(pa, bv, acc[nt]);
    }
  }

  float inv[8];
#pragma unroll
  for (int j = 0; j < 8; ++j) inv[j] = 1.0f / l_run[j];
#pragma unroll
  for (int nt = 0; nt < 4; ++nt) {
    int col = h * DK + nt * 16 + (lane & 15);
#pragma unroll
    for (int j = 0; j < 8; ++j) {
      int row = q0 + j + 8 * g;
      Ao[((size_t)(b * S + row)) * D + col] = f2bf(acc[nt][j] * inv[j]);
    }
  }
}

// ---- kernel 3: out = Ao @ Wo^T + bo (bf16 in, fp32 out), 32M x 64N per wave
__global__ __launch_bounds__(256) void proj_out_kernel(
    const bf16_t* __restrict__ Ao, const bf16_t* __restrict__ Wo,
    const float* __restrict__ bo, float* __restrict__ out) {
  const int lane = threadIdx.x & 31;
  const int wgid = blockIdx.x * (blockDim.x >> 5) + (threadIdx.x >> 5);
  const int m0 = (wgid >> 4) * 32;
  const int n0 = (wgid & 15) * 64;
  const int g = lane >> 4;

  v8f acc0[4] = {vzero8(), vzero8(), vzero8(), vzero8()};
  v8f acc1[4] = {vzero8(), vzero8(), vzero8(), vzero8()};
  for (int k0 = 0; k0 < D; k0 += 32) {
    if (k0 + 32 < D) {
      __builtin_prefetch(Ao + (size_t)(m0 + (lane & 31)) * D + k0 + 32, 0, 3);
      __builtin_prefetch(Wo + (size_t)(n0 + (lane & 31)) * D + k0 + 32, 0, 3);
    }
    v16bf a0 = load_a_bf(Ao, D, m0, k0, lane);
    v16bf a1 = load_a_bf(Ao, D, m0 + 16, k0, lane);
#pragma unroll
    for (int nt = 0; nt < 4; ++nt) {
      v16bf bw = load_b_bf(Wo, D, n0 + nt * 16, k0, lane);
      acc0[nt] = WMMA_BF16(a0, bw, acc0[nt]);
      acc1[nt] = WMMA_BF16(a1, bw, acc1[nt]);
    }
  }
#pragma unroll
  for (int nt = 0; nt < 4; ++nt) {
    int n = n0 + nt * 16 + (lane & 15);
    float bval = bo[n];
#pragma unroll
    for (int j = 0; j < 8; ++j) {
      int m = m0 + j + 8 * g;
      out[(size_t)m * D + n] = acc0[nt][j] + bval;
    }
#pragma unroll
    for (int j = 0; j < 8; ++j) {
      int m = m0 + 16 + j + 8 * g;
      out[(size_t)m * D + n] = acc1[nt][j] + bval;
    }
  }
}

extern "C" void kernel_launch(void* const* d_in, const int* in_sizes, int n_in,
                              void* d_out, int out_size, void* d_ws, size_t ws_size,
                              hipStream_t stream) {
  const float* q  = (const float*)d_in[0];
  const float* k  = (const float*)d_in[1];
  const float* v  = (const float*)d_in[2];
  const float* Wq = (const float*)d_in[3];
  const float* bq = (const float*)d_in[4];
  const float* Wk = (const float*)d_in[5];
  const float* bk = (const float*)d_in[6];
  const float* Wv = (const float*)d_in[7];
  const float* bv = (const float*)d_in[8];
  const float* Wo = (const float*)d_in[9];
  const float* bo = (const float*)d_in[10];
  float* out = (float*)d_out;

  const size_t NX = (size_t)MT * D;   // 4M elements
  const size_t NW = (size_t)D * D;    // 1M elements
  bf16_t* Xq  = (bf16_t*)d_ws;        // bf16 copies of inputs
  bf16_t* Xk  = Xq + NX;
  bf16_t* Xv  = Xk + NX;
  bf16_t* Wqb = Xv + NX;
  bf16_t* Wkb = Wqb + NW;
  bf16_t* Wvb = Wkb + NW;
  bf16_t* Wob = Wvb + NW;
  bf16_t* Qh  = Wob + NW;             // [B,H,S,DK]
  bf16_t* Kh  = Qh + NX;              // [B,H,S,DK]
  bf16_t* Vt  = Kh + NX;              // [B,H,DK,S]
  bf16_t* Ao  = Vt + NX;              // [B,S,D]

  // one-time fp32 -> bf16 conversion (2048 elems per block)
  cvt_bf16_kernel<<<dim3((unsigned)(NX / 2048)), dim3(256), 0, stream>>>(q, Xq);
  cvt_bf16_kernel<<<dim3((unsigned)(NX / 2048)), dim3(256), 0, stream>>>(k, Xk);
  cvt_bf16_kernel<<<dim3((unsigned)(NX / 2048)), dim3(256), 0, stream>>>(v, Xv);
  cvt_bf16_kernel<<<dim3((unsigned)(NW / 2048)), dim3(256), 0, stream>>>(Wq, Wqb);
  cvt_bf16_kernel<<<dim3((unsigned)(NW / 2048)), dim3(256), 0, stream>>>(Wk, Wkb);
  cvt_bf16_kernel<<<dim3((unsigned)(NW / 2048)), dim3(256), 0, stream>>>(Wv, Wvb);
  cvt_bf16_kernel<<<dim3((unsigned)(NW / 2048)), dim3(256), 0, stream>>>(Wo, Wob);

  // projections: 2048 wave-tiles (128 m-tiles x 16 n-groups), 8 waves/block
  dim3 pb(256), pg(256);
  proj_qkv_kernel<<<pg, pb, 0, stream>>>(Xq, Wqb, bq, Qh, 0);
  proj_qkv_kernel<<<pg, pb, 0, stream>>>(Xk, Wkb, bk, Kh, 0);
  proj_qkv_kernel<<<pg, pb, 0, stream>>>(Xv, Wvb, bv, Vt, 1);
  attn_kernel<<<dim3(1024), dim3(128), 0, stream>>>(Qh, Kh, Vt, Ao);
  proj_out_kernel<<<pg, pb, 0, stream>>>(Ao, Wob, bo, out);
}
